// ViViT_6270652252417
// MI455X (gfx1250) — matrix operand order repaired and checked by reading, not compiled, not run on hardware
//
#include <hip/hip_runtime.h>
#include <hip/hip_bf16.h>
#include <math.h>

// ---------------------------------------------------------------------------
// ViViT factorized space-time transformer for MI455X (gfx1250, wave32, WMMA).
// GEMMs run on v_wmma_f32_16x16x32_bf16 with f32 accumulation, with block
// tiles staged global->LDS via global_load_async_to_lds_b128 (ASYNCcnt path).
// ---------------------------------------------------------------------------

typedef __bf16 bf16_t;
typedef __attribute__((ext_vector_type(16))) __bf16 v16bf;
typedef __attribute__((ext_vector_type(8)))  float  v8f;
typedef __attribute__((ext_vector_type(4)))  int    v4i_t;

#define E_    512
#define I_    512
#define F_    2048
#define NF_   3584      // 3I + F
#define H_    8
#define D_    64
#define T_    60
#define P_    256
#define NTOK  30720     // B*T*P
#define CHUNK 7680      // tokens per pipeline chunk (30 spatial / 128 temporal seqs)

#define BM 128
#define BN 64
#define BK 32

#if __has_builtin(__builtin_amdgcn_global_load_async_to_lds_b128)
#define HAVE_ASYNC_LDS 1
#else
#define HAVE_ASYNC_LDS 0
#endif

// 16-byte global -> LDS copy: async (ASYNCcnt) when available, else reg relay.
__device__ __forceinline__ void cp16(const bf16_t* g, bf16_t* l) {
#if HAVE_ASYNC_LDS
  __builtin_amdgcn_global_load_async_to_lds_b128(
      (__attribute__((address_space(1))) v4i_t*)g,
      (__attribute__((address_space(3))) v4i_t*)l, 0, 0);
#else
  *(uint4*)l = *(const uint4*)g;
#endif
}
__device__ __forceinline__ void cp_wait() {
#if HAVE_ASYNC_LDS
#if __has_builtin(__builtin_amdgcn_s_wait_asynccnt)
  __builtin_amdgcn_s_wait_asynccnt(0);
#else
  asm volatile("s_wait_asynccnt 0" ::: "memory");
#endif
#endif
}

// Map local chunk token index -> element offset of that token's row in Y[B,T,P,E].
__device__ __forceinline__ size_t map_off(int l, int temporal, int base) {
  if (!temporal) return ((size_t)(base + l)) << 9;
  int ls  = l / 60;
  int pos = l - ls * 60;
  int seq = base + ls;          // seq = b*P + p
  int b   = seq >> 8;
  int p   = seq & 255;
  return ((size_t)((b * 60 + pos) * 256 + p)) << 9;
}

// A fragment 16x32 bf16 (row-major, stride lda). ISA 7.12.2: lanes 0-15 carry
// K[0..7]+K[16..23] of row=lane; lanes 16-31 carry K[8..15]+K[24..31].
__device__ __forceinline__ v16bf frag_a(const bf16_t* A, int lda, int tm, int k0,
                                        int lane, int maxr) {
  int r = tm + (lane & 15); r = r < maxr ? r : maxr - 1;
  const bf16_t* p = A + (size_t)r * lda + k0 + ((lane >> 4) << 3);
  v16bf f;
  ((uint4*)&f)[0] = *(const uint4*)p;
  ((uint4*)&f)[1] = *(const uint4*)(p + 16);
  return f;
}

// B fragment 32x16 bf16 from B^T rows: lanes 0-15 K[0..7]+K[16..23], lanes 16-31 K[8..15]+K[24..31].
__device__ __forceinline__ v16bf frag_b(const bf16_t* BT, int ldb, int tn, int k0,
                                        int lane, int maxr) {
  int n = tn + (lane & 15); n = n < maxr ? n : maxr - 1;
  const bf16_t* p = BT + (size_t)n * ldb + k0 + ((lane >> 4) << 3);
  v16bf f;
  ((uint4*)&f)[0] = *(const uint4*)p;
  ((uint4*)&f)[1] = *(const uint4*)(p + 16);
  return f;
}

__device__ __forceinline__ v8f wmma_bf16(v16bf a, v16bf b, v8f c) {
  return __builtin_amdgcn_wmma_f32_16x16x32_bf16(false, a, false, b, (short)0, c,
                                                 false, false);
}

// ---------------------------------------------------------------------------
// GEMM: C[M,N] = A[M,K](bf16) * B given as BT[N,K](bf16).
// Block = 256 thr (8 waves), tile 128x64, K-step 32 staged in LDS.
// Each wave: one A frag reused over 4 B frags -> 4 WMMAs / K-step.
// MODE 0: f32   MODE 1: bf16   MODE 2: exact GELU -> bf16
// MODE 3: Y[token] += acc + C  (residual scatter-add epilogue)
// ---------------------------------------------------------------------------
template <int MODE>
__global__ __launch_bounds__(256) void gemm_bf16_k(
    const bf16_t* __restrict__ A, int lda,
    const bf16_t* __restrict__ BT, int ldb,
    int M, int N, int K,
    float* __restrict__ Cf, bf16_t* __restrict__ Cb, int ldc,
    const float* __restrict__ acc, float* __restrict__ Y,
    int temporal, int base)
{
  __shared__ __align__(16) bf16_t sA[BM * BK];   // 8 KB
  __shared__ __align__(16) bf16_t sB[BN * BK];   // 4 KB
  int tid  = threadIdx.x;
  int lane = tid & 31;
  int wave = tid >> 5;
  int tn0 = blockIdx.x * BN;
  int tm0 = blockIdx.y * BM;

  // staging map: 16B chunks; A has 512 chunks (2/thread), B has 256 (1/thread)
  int ca0 = tid * 2, ca1 = tid * 2 + 1;
  int ar0 = ca0 >> 2, ac0 = (ca0 & 3) << 3;
  int ar1 = ca1 >> 2, ac1 = (ca1 & 3) << 3;
  int br  = tid >> 2, bc  = (tid & 3) << 3;

  v8f accv[4] = {};
  for (int k0 = 0; k0 < K; k0 += BK) {
    __syncthreads();   // previous iteration's readers done before overwrite
    cp16(A  + (size_t)(tm0 + ar0) * lda + k0 + ac0, &sA[ar0 * BK + ac0]);
    cp16(A  + (size_t)(tm0 + ar1) * lda + k0 + ac1, &sA[ar1 * BK + ac1]);
    cp16(BT + (size_t)(tn0 + br ) * ldb + k0 + bc,  &sB[br  * BK + bc ]);
    cp_wait();
    __syncthreads();

    v16bf a;
    {
      const bf16_t* p = &sA[(wave * 16 + (lane & 15)) * BK + ((lane >> 4) << 3)];
      ((uint4*)&a)[0] = *(const uint4*)p;
      ((uint4*)&a)[1] = *(const uint4*)(p + 16);
    }
#pragma unroll
    for (int j = 0; j < 4; ++j) {
      v16bf b;
      const bf16_t* p = &sB[(j * 16 + (lane & 15)) * BK + ((lane >> 4) << 3)];
      ((uint4*)&b)[0] = *(const uint4*)p;
      ((uint4*)&b)[1] = *(const uint4*)(p + 16);
      accv[j] = wmma_bf16(a, b, accv[j]);
    }
  }

  int rbl = tm0 + wave * 16 + ((lane >> 4) << 3);
#pragma unroll
  for (int j = 0; j < 4; ++j) {
    int col = tn0 + j * 16 + (lane & 15);
#pragma unroll
    for (int jj = 0; jj < 8; ++jj) {
      int row = rbl + jj;
      float v = accv[j][jj];
      if (MODE == 0) {
        Cf[(size_t)row * ldc + col] = v;
      } else if (MODE == 1) {
        Cb[(size_t)row * ldc + col] = (bf16_t)v;
      } else if (MODE == 2) {
        float gl = 0.5f * v * (1.0f + erff(v * 0.70710678118f));
        Cb[(size_t)row * ldc + col] = (bf16_t)gl;
      } else {
        size_t off = map_off(row, temporal, base);
        Y[off + col] += acc[(size_t)row * I_ + col] + v;
      }
    }
  }
}

// ---------------------------------------------------------------------------
// LayerNorm over E=512 of gathered token rows -> bf16 (GEMM A operand).
// ---------------------------------------------------------------------------
__global__ __launch_bounds__(128) void ln_gather_k(
    const float* __restrict__ Y, const float* __restrict__ g,
    const float* __restrict__ b, bf16_t* __restrict__ Hout,
    int temporal, int base)
{
  __shared__ float xs[E_];
  __shared__ float r1[4], r2[4];
  int l = blockIdx.x;
  size_t off = map_off(l, temporal, base);
  int tid = threadIdx.x;
  float s = 0.f, s2 = 0.f;
#pragma unroll
  for (int e = 0; e < 4; ++e) {
    float v = Y[off + tid * 4 + e];
    xs[tid * 4 + e] = v;
    s += v; s2 += v * v;
  }
  for (int o = 16; o > 0; o >>= 1) { s += __shfl_xor(s, o, 32); s2 += __shfl_xor(s2, o, 32); }
  if ((tid & 31) == 0) { r1[tid >> 5] = s; r2[tid >> 5] = s2; }
  __syncthreads();
  float mean = (r1[0] + r1[1] + r1[2] + r1[3]) * (1.0f / E_);
  float ms   = (r2[0] + r2[1] + r2[2] + r2[3]) * (1.0f / E_);
  float rstd = rsqrtf(ms - mean * mean + 1e-5f);
#pragma unroll
  for (int e = 0; e < 4; ++e) {
    int cdx = tid * 4 + e;
    float v = (xs[cdx] - mean) * rstd * g[cdx] + b[cdx];
    Hout[(size_t)l * E_ + cdx] = (bf16_t)v;
  }
}

// ---------------------------------------------------------------------------
// q/k LayerNorm (over I=512) + per-head RoPE (D=64, half-rotation) -> bf16.
// ---------------------------------------------------------------------------
__global__ __launch_bounds__(128) void ln_rope_k(
    const float* __restrict__ Qin, const float* __restrict__ g,
    const float* __restrict__ b, const float* __restrict__ ct,
    const float* __restrict__ st, bf16_t* __restrict__ Qout, int npos)
{
  __shared__ float xs[I_];
  __shared__ float r1[4], r2[4];
  int l = blockIdx.x;
  int pos = l % npos;
  int tid = threadIdx.x;
  float s = 0.f, s2 = 0.f;
#pragma unroll
  for (int e = 0; e < 4; ++e) {
    float v = Qin[(size_t)l * I_ + tid * 4 + e];
    xs[tid * 4 + e] = v;
    s += v; s2 += v * v;
  }
  for (int o = 16; o > 0; o >>= 1) { s += __shfl_xor(s, o, 32); s2 += __shfl_xor(s2, o, 32); }
  if ((tid & 31) == 0) { r1[tid >> 5] = s; r2[tid >> 5] = s2; }
  __syncthreads();
  float mean = (r1[0] + r1[1] + r1[2] + r1[3]) * (1.0f / I_);
  float ms   = (r2[0] + r2[1] + r2[2] + r2[3]) * (1.0f / I_);
  float rstd = rsqrtf(ms - mean * mean + 1e-5f);
#pragma unroll
  for (int e = 0; e < 4; ++e) {
    int cdx = tid * 4 + e;
    xs[cdx] = (xs[cdx] - mean) * rstd * g[cdx] + b[cdx];
  }
  __syncthreads();
#pragma unroll
  for (int e = 0; e < 4; ++e) {
    int cdx = tid * 4 + e;
    int d = cdx & 63;
    float xv  = xs[cdx];
    float rot = (d < 32) ? -xs[cdx + 32] : xs[cdx - 32];
    float ov  = xv * ct[pos * 64 + d] + rot * st[pos * 64 + d];
    Qout[(size_t)l * I_ + cdx] = (bf16_t)ov;
  }
}

// ---------------------------------------------------------------------------
// Attention for one (sequence, head): S = (Q K^T)/8 in LDS f32 (NPxNP),
// wave-parallel softmax, O = A V via WMMA with V^T staged in LDS.
// NP=256 -> 295KB LDS (one WG per 320KB WGP); NP=64 -> 24KB.
// ---------------------------------------------------------------------------
__global__ __launch_bounds__(256) void attn_k(
    const bf16_t* __restrict__ qbuf, const bf16_t* __restrict__ kbuf,
    const bf16_t* __restrict__ vbuf, bf16_t* __restrict__ obuf,
    int N, int NP)
{
  extern __shared__ char smem[];
  float*  sS  = (float*)smem;                               // NP x NP scores
  bf16_t* sVT = (bf16_t*)(smem + (size_t)NP * NP * 4);      // 64 x NP  V^T
  int tid = threadIdx.x, lane = tid & 31, wave = tid >> 5;
  int seq = blockIdx.x >> 3, head = blockIdx.x & 7;
  size_t bp = ((size_t)seq * N) * I_ + head * D_;
  const bf16_t* q = qbuf + bp;
  const bf16_t* k = kbuf + bp;
  const bf16_t* v = vbuf + bp;
  bf16_t*       o = obuf + bp;

  for (int idx = tid; idx < D_ * NP; idx += 256) {
    int d = idx / NP, m = idx - d * NP;
    sVT[idx] = (m < N) ? v[(size_t)m * I_ + d] : (bf16_t)0.0f;
  }
  __syncthreads();

  int nt = NP >> 4;
  for (int t = wave; t < nt * nt; t += 8) {
    int ti = t / nt;
    int tm = ti << 4, tn = (t - ti * nt) << 4;
    v8f c = {};
#pragma unroll
    for (int k0 = 0; k0 < D_; k0 += 32) {
      v16bf a = frag_a(q, I_, tm, k0, lane, N);
      v16bf b = frag_b(k, I_, tn, k0, lane, N);
      c = wmma_bf16(a, b, c);
    }
    int col = tn + (lane & 15);
    int rb  = tm + ((lane >> 4) << 3);
#pragma unroll
    for (int j = 0; j < 8; ++j)
      sS[(size_t)(rb + j) * NP + col] = c[j] * 0.125f;
  }
  __syncthreads();

  for (int row = wave; row < NP; row += 8) {
    float* r = sS + (size_t)row * NP;
    float mx = -3.0e38f;
    for (int cc = lane; cc < N; cc += 32) mx = fmaxf(mx, r[cc]);
    for (int off = 16; off > 0; off >>= 1) mx = fmaxf(mx, __shfl_xor(mx, off, 32));
    float sum = 0.f;
    for (int cc = lane; cc < NP; cc += 32) {
      float e = (cc < N) ? __expf(r[cc] - mx) : 0.0f;
      r[cc] = e;
      sum += e;
    }
    for (int off = 16; off > 0; off >>= 1) sum += __shfl_xor(sum, off, 32);
    float inv = 1.0f / sum;
    for (int cc = lane; cc < NP; cc += 32) r[cc] *= inv;
  }
  __syncthreads();

  for (int t = wave; t < nt * 4; t += 8) {
    int tm = (t >> 2) << 4, tn = (t & 3) << 4;
    v8f c = {};
    for (int k0 = 0; k0 < NP; k0 += 32) {
      v16bf a;
      {
        int r = tm + (lane & 15);
        const float* ap = sS + (size_t)r * NP + k0 + ((lane >> 4) << 3);
#pragma unroll
        for (int e = 0; e < 8; ++e) { a[e] = (bf16_t)ap[e]; a[e + 8] = (bf16_t)ap[e + 16]; }
      }
      v16bf b;
      {
        int n = tn + (lane & 15);
        const bf16_t* bpd = sVT + (size_t)n * NP + k0 + ((lane >> 4) << 3);
        ((uint4*)&b)[0] = *(const uint4*)bpd;
        ((uint4*)&b)[1] = *(const uint4*)(bpd + 16);
      }
      c = wmma_bf16(a, b, c);
    }
    int col = tn + (lane & 15);
    int rb  = tm + ((lane >> 4) << 3);
#pragma unroll
    for (int j = 0; j < 8; ++j) {
      int row = rb + j;
      if (row < N) o[(size_t)row * I_ + col] = (bf16_t)c[j];
    }
  }
}

__global__ void rope_init_k(float* ct, float* st) {
  int n = blockIdx.x, d = threadIdx.x;  // 256 x 64
  float inv = powf(10000.0f, -(float)(2 * (d & 31)) / 64.0f);
  float ang = (float)n * inv;
  ct[n * 64 + d] = cosf(ang);
  st[n * 64 + d] = sinf(ang);
}

__global__ void transpose_bf16_k(const float* __restrict__ in,
                                 bf16_t* __restrict__ out, int R, int C) {
  int idx = blockIdx.x * 256 + threadIdx.x;
  if (idx < R * C) {
    int r = idx / C, cc = idx - r * C;
    out[(size_t)cc * R + r] = (bf16_t)in[idx];
  }
}

// ---------------------------------------------------------------------------
extern "C" void kernel_launch(void* const* d_in, const int* in_sizes, int n_in,
                              void* d_out, int out_size, void* d_ws, size_t ws_size,
                              hipStream_t stream)
{
  (void)in_sizes; (void)n_in; (void)out_size; (void)ws_size;
  const float* x = (const float*)d_in[0];
  const float* Wf_[2]  = {(const float*)d_in[1],  (const float*)d_in[10]};
  const float* Wo_[2]  = {(const float*)d_in[2],  (const float*)d_in[11]};
  const float* Wff_[2] = {(const float*)d_in[3],  (const float*)d_in[12]};
  const float* g_[2]   = {(const float*)d_in[4],  (const float*)d_in[13]};
  const float* bb_[2]  = {(const float*)d_in[5],  (const float*)d_in[14]};
  const float* qg_[2]  = {(const float*)d_in[6],  (const float*)d_in[15]};
  const float* qb_[2]  = {(const float*)d_in[7],  (const float*)d_in[16]};
  const float* kg_[2]  = {(const float*)d_in[8],  (const float*)d_in[17]};
  const float* kb_[2]  = {(const float*)d_in[9],  (const float*)d_in[18]};

  char* w = (char*)d_ws;
  auto alloc = [&](size_t bytes) -> void* {
    void* p = (void*)w; w += (bytes + 255) & ~(size_t)255; return p;
  };
  float*  Y    = (float*) alloc((size_t)NTOK * E_ * 4);
  bf16_t* WfT  = (bf16_t*)alloc((size_t)NF_ * E_ * 2);
  bf16_t* WoT  = (bf16_t*)alloc((size_t)E_ * I_ * 2);
  bf16_t* WffT = (bf16_t*)alloc((size_t)E_ * F_ * 2);
  bf16_t* Hb   = (bf16_t*)alloc((size_t)CHUNK * E_ * 2);
  float*  Qf   = (float*) alloc((size_t)CHUNK * I_ * 4);
  float*  Kf   = (float*) alloc((size_t)CHUNK * I_ * 4);
  float*  Acc  = (float*) alloc((size_t)CHUNK * E_ * 4);
  bf16_t* Qb   = (bf16_t*)alloc((size_t)CHUNK * I_ * 2);
  bf16_t* Kb   = (bf16_t*)alloc((size_t)CHUNK * I_ * 2);
  bf16_t* Vb   = (bf16_t*)alloc((size_t)CHUNK * I_ * 2);
  bf16_t* FFb  = (bf16_t*)alloc((size_t)CHUNK * F_ * 2);
  bf16_t* Ob   = (bf16_t*)alloc((size_t)CHUNK * I_ * 2);
  float*  Ct   = (float*) alloc(256 * 64 * 4);
  float*  St   = (float*) alloc(256 * 64 * 4);

  (void)hipMemcpyAsync(Y, x, (size_t)NTOK * E_ * 4, hipMemcpyDeviceToDevice, stream);
  rope_init_k<<<256, 64, 0, stream>>>(Ct, St);

  for (int stage = 0; stage < 2; ++stage) {
    int temporal = stage;
    int N  = temporal ? 60 : 256;
    int NP = temporal ? 64 : 256;
    int seqs_per_chunk = temporal ? 128 : 30;
    size_t attn_lds = (size_t)NP * NP * 4 + (size_t)D_ * NP * 2;

    for (int blk = 0; blk < 4; ++blk) {
      const float* Wf  = Wf_[stage]  + (size_t)blk * E_ * NF_;
      const float* Wo  = Wo_[stage]  + (size_t)blk * I_ * E_;
      const float* Wff = Wff_[stage] + (size_t)blk * F_ * E_;
      const float* gg  = g_[stage]   + (size_t)blk * E_;
      const float* gb  = bb_[stage]  + (size_t)blk * E_;
      const float* qgg = qg_[stage]  + (size_t)blk * I_;
      const float* qgb = qb_[stage]  + (size_t)blk * I_;
      const float* kgg = kg_[stage]  + (size_t)blk * I_;
      const float* kgb = kb_[stage]  + (size_t)blk * I_;

      transpose_bf16_k<<<(E_ * NF_ + 255) / 256, 256, 0, stream>>>(Wf,  WfT,  E_, NF_);
      transpose_bf16_k<<<(I_ * E_  + 255) / 256, 256, 0, stream>>>(Wo,  WoT,  I_, E_);
      transpose_bf16_k<<<(F_ * E_  + 255) / 256, 256, 0, stream>>>(Wff, WffT, F_, E_);

      for (int c = 0; c < 4; ++c) {
        int base = temporal ? c * seqs_per_chunk : c * CHUNK;

        ln_gather_k<<<CHUNK, 128, 0, stream>>>(Y, gg, gb, Hb, temporal, base);

        dim3 gq(E_ / BN, CHUNK / BM);    // 8 x 60
        dim3 gff(F_ / BN, CHUNK / BM);   // 32 x 60
        // fused projection, column-sliced: q | k | v | ff(+GELU)
        gemm_bf16_k<0><<<gq, 256, 0, stream>>>(Hb, E_, WfT,                      E_,
            CHUNK, I_, E_, Qf, nullptr, I_, nullptr, nullptr, 0, 0);
        gemm_bf16_k<0><<<gq, 256, 0, stream>>>(Hb, E_, WfT + (size_t)I_ * E_,    E_,
            CHUNK, I_, E_, Kf, nullptr, I_, nullptr, nullptr, 0, 0);
        gemm_bf16_k<1><<<gq, 256, 0, stream>>>(Hb, E_, WfT + (size_t)2*I_ * E_,  E_,
            CHUNK, I_, E_, nullptr, Vb, I_, nullptr, nullptr, 0, 0);
        gemm_bf16_k<2><<<gff, 256, 0, stream>>>(Hb, E_, WfT + (size_t)3*I_ * E_, E_,
            CHUNK, F_, E_, nullptr, FFb, F_, nullptr, nullptr, 0, 0);

        ln_rope_k<<<CHUNK, 128, 0, stream>>>(Qf, qgg, qgb, Ct, St, Qb, N);
        ln_rope_k<<<CHUNK, 128, 0, stream>>>(Kf, kgg, kgb, Ct, St, Kb, N);

        attn_k<<<seqs_per_chunk * H_, 256, attn_lds, stream>>>(Qb, Kb, Vb, Ob, N, NP);

        gemm_bf16_k<0><<<gq, 256, 0, stream>>>(Ob, I_, WoT, I_,
            CHUNK, E_, I_, Acc, nullptr, E_, nullptr, nullptr, 0, 0);
        gemm_bf16_k<3><<<gq, 256, 0, stream>>>(FFb, F_, WffT, F_,
            CHUNK, E_, F_, nullptr, nullptr, E_, Acc, Y, temporal, base);
      }
    }
  }
  (void)hipMemcpyAsync(d_out, Y, (size_t)NTOK * E_ * 4, hipMemcpyDeviceToDevice, stream);
}
